// LongContextAttention_91276644974766
// MI455X (gfx1250) — matrix operand order repaired and checked by reading, not compiled, not run on hardware
//
#include <hip/hip_runtime.h>
#include <hip/hip_bf16.h>

typedef __attribute__((ext_vector_type(16))) __bf16 v16bf;
typedef __attribute__((ext_vector_type(8)))  __bf16 v8bf;
typedef __attribute__((ext_vector_type(2)))  __bf16 v2bf;
typedef __attribute__((ext_vector_type(8)))  float  v8f;

namespace {
constexpr int  Bb = 2;
constexpr int  Ss = 2048;
constexpr int  Hh = 16;
constexpr int  Dd = 128;
constexpr int  BK = 32;    // KV tile rows per iteration
constexpr int  BM = 128;   // Q rows per workgroup (4 waves x 32 rows)
// 1/sqrt(128) * log2(e): scores computed in log2 domain so softmax uses raw v_exp_f32
constexpr float SCALE_LOG2E = 0.08838834764831845f * 1.4426950408889634f;
}

__device__ __forceinline__ v8bf cvt8(float4 a, float4 b, float s) {
  v8bf r;
  r[0] = (__bf16)(a.x * s); r[1] = (__bf16)(a.y * s);
  r[2] = (__bf16)(a.z * s); r[3] = (__bf16)(a.w * s);
  r[4] = (__bf16)(b.x * s); r[5] = (__bf16)(b.y * s);
  r[6] = (__bf16)(b.z * s); r[7] = (__bf16)(b.w * s);
  return r;
}

__device__ __forceinline__ v16bf cat16(v8bf lo, v8bf hi) {
  return __builtin_shufflevector(lo, hi,
      0,1,2,3,4,5,6,7,8,9,10,11,12,13,14,15);
}

// 16 contiguous bf16 from LDS (two ds_load_b128)
__device__ __forceinline__ v16bf lds_frag(const __bf16* p) {
  v8bf lo = *(const v8bf*)p;
  v8bf hi = *(const v8bf*)(p + 8);
  return cat16(lo, hi);
}

// two disjoint 8-element groups (A-fragment K-halves)
__device__ __forceinline__ v16bf lds_frag2(const __bf16* p0, const __bf16* p1) {
  return cat16(*(const v8bf*)p0, *(const v8bf*)p1);
}

// ---- DPP16 row-rotate (within 16-lane row; wave32 = two rows) ----
template <int CTRL>
__device__ __forceinline__ float fdpp(float x) {
  int i = __float_as_int(x);
  i = __builtin_amdgcn_update_dpp(i, i, CTRL, 0xf, 0xf, false);
  return __int_as_float(i);
}

// all-reduce across the 16-lane row via rotate-combine ladder (pure VALU)
__device__ __forceinline__ float rmax16(float x) {
  x = fmaxf(x, fdpp<0x121>(x));  // row_ror:1
  x = fmaxf(x, fdpp<0x122>(x));  // row_ror:2
  x = fmaxf(x, fdpp<0x124>(x));  // row_ror:4
  x = fmaxf(x, fdpp<0x128>(x));  // row_ror:8
  return x;
}

__device__ __forceinline__ float rsum16(float x) {
  x += fdpp<0x121>(x);
  x += fdpp<0x122>(x);
  x += fdpp<0x124>(x);
  x += fdpp<0x128>(x);
  return x;
}

__global__ __launch_bounds__(128)
void fa_bf16_wmma_kernel(const float* __restrict__ q,
                         const float* __restrict__ k,
                         const float* __restrict__ v,
                         float* __restrict__ out) {
  // K-dim of the PV GEMM is stored in permuted "slot" order:
  //   slot(c) = (c & 15)*2 + (c >> 4)
  // applied identically to P columns and V rows, so p0/p1 pack into one b32
  // store while all fragment *read* addresses stay contiguous/unchanged.
  __shared__ __bf16 sK[2][BK][Dd];   // K tiles, ping-pong            (16 KB)
  __shared__ __bf16 sVt[2][Dd][BK];  // V^T tiles, k in slot order    (16 KB)
  __shared__ __bf16 sP[4][32][BK];   // per-wave P staging, slot order ( 8 KB)

  const int tid  = threadIdx.x;
  const int wave = tid >> 5;        // 0..3
  const int lane = tid & 31;
  const int ln16 = lane & 15;
  const int sel  = lane >> 4;       // 0: lanes 0-15, 1: lanes 16-31

  constexpr int qtiles = Ss / BM;              // 16
  const int wg = blockIdx.x;
  const int qt = wg % qtiles;
  const int h  = (wg / qtiles) % Hh;
  const int b  = wg / (qtiles * Hh);

  const int q0 = qt * BM + wave * 32;          // first Q row of this wave (32 rows)

  // ---- Q fragments: 2 M-tiles x 4 K-chunks (16-bit A layout), scale*log2e folded
  v16bf qa[2][4];
  #pragma unroll
  for (int mt = 0; mt < 2; ++mt) {
    const float* qp = q + (((size_t)b * Ss + q0 + mt * 16 + ln16) * Hh + h) * Dd;
    #pragma unroll
    for (int c = 0; c < 4; ++c) {
      const float* p0 = qp + 32 * c + sel * 8;        // -> frag elems 0..7
      const float* p1 = qp + 32 * c + 16 + sel * 8;   // -> frag elems 8..15
      float4 a0 = *(const float4*)(p0);
      float4 a1 = *(const float4*)(p0 + 4);
      float4 b0 = *(const float4*)(p1);
      float4 b1 = *(const float4*)(p1 + 4);
      qa[mt][c] = cat16(cvt8(a0, a1, SCALE_LOG2E), cvt8(b0, b1, SCALE_LOG2E));
    }
  }

  v8f o[2][8] = {};              // O accumulators: 2 M-tiles x 8 D-tiles
  float mrow[2][8], lrow[2][8];  // per-row running max / sum
  #pragma unroll
  for (int mt = 0; mt < 2; ++mt)
    #pragma unroll
    for (int r = 0; r < 8; ++r) { mrow[mt][r] = -1e30f; lrow[mt][r] = 0.f; }

  // cooperative tile-load mapping: 128 threads cover 32x128 elements
  const int r32   = tid >> 2;          // 0..31 : tile row
  const int c32   = (tid & 3) << 5;    // 0,32,64,96 : col group of 32
  const int slotr = (r32 & 15) * 2 + (r32 >> 4);  // V row -> k-slot

  int buf = 0;
  for (int s0 = 0; s0 < Ss; s0 += BK, buf ^= 1) {
    const size_t base = (((size_t)b * Ss + s0 + r32) * Hh + h) * (size_t)Dd + c32;
    const float* kp = k + base;
    const float* vp = v + base;
    if (s0 + BK < Ss) {  // speculative prefetch of next tile -> global_prefetch_b8
      __builtin_prefetch(kp + (size_t)BK * Hh * Dd, 0, 0);
      __builtin_prefetch(vp + (size_t)BK * Hh * Dd, 0, 0);
    }

    // K: row-major bf16, packed b128 stores
    #pragma unroll
    for (int ch = 0; ch < 4; ++ch) {
      float4 a = *(const float4*)(kp + ch * 8);
      float4 d = *(const float4*)(kp + ch * 8 + 4);
      *(v8bf*)&sK[buf][r32][c32 + ch * 8] = cvt8(a, d, 1.f);
    }
    // V: transposed scatter into k-slot order (b16 stores)
    #pragma unroll
    for (int ch = 0; ch < 4; ++ch) {
      float4 a = *(const float4*)(vp + ch * 8);
      float4 d = *(const float4*)(vp + ch * 8 + 4);
      v8bf t8 = cvt8(a, d, 1.f);
      #pragma unroll
      for (int j = 0; j < 8; ++j)
        sVt[buf][c32 + ch * 8 + j][slotr] = t8[j];
    }
    __syncthreads();     // single barrier per iteration: tile `buf` visible

    // ---- scores: 2 M-tiles x 2 N-tiles; each K fragment feeds 2 WMMAs ----
    v8f sc[2][2];        // [mt][t]
    #pragma unroll
    for (int mt = 0; mt < 2; ++mt)
      #pragma unroll
      for (int t = 0; t < 2; ++t) sc[mt][t] = (v8f){};
    #pragma unroll
    for (int t = 0; t < 2; ++t) {
      #pragma unroll
      for (int c = 0; c < 4; ++c) {
        v16bf bf = lds_frag2(&sK[buf][t * 16 + ln16][32 * c + sel * 8],
                             &sK[buf][t * 16 + ln16][32 * c + 16 + sel * 8]);
        sc[0][t] = __builtin_amdgcn_wmma_f32_16x16x32_bf16(
                       false, qa[0][c], false, bf, (short)0, sc[0][t], false, false);
        sc[1][t] = __builtin_amdgcn_wmma_f32_16x16x32_bf16(
                       false, qa[1][c], false, bf, (short)0, sc[1][t], false, false);
      }
    }

    // ---- online softmax in log2 domain (joint max over both N-tiles) ----
    #pragma unroll
    for (int mt = 0; mt < 2; ++mt) {
      #pragma unroll
      for (int r = 0; r < 8; ++r) {
        float s0v = sc[mt][0][r];
        float s1v = sc[mt][1][r];
        float mx = rmax16(fmaxf(s0v, s1v));
        float mn   = fmaxf(mrow[mt][r], mx);
        float p0   = exp2f(s0v - mn);
        float p1   = exp2f(s1v - mn);
        float corr = exp2f(mrow[mt][r] - mn);
        mrow[mt][r] = mn;
        float rs = rsum16(p0 + p1);
        lrow[mt][r] = lrow[mt][r] * corr + rs;
        #pragma unroll
        for (int dt = 0; dt < 8; ++dt) o[mt][dt][r] *= corr;
        // stage P: cols ln16 (t=0) and 16+ln16 (t=1) are adjacent slots
        // 2*ln16, 2*ln16+1 -> single packed b32 store
        v2bf pk; pk[0] = (__bf16)p0; pk[1] = (__bf16)p1;
        *(v2bf*)&sP[wave][mt * 16 + r + sel * 8][2 * ln16] = pk;
      }
    }

    // ---- O += P V : each V fragment feeds 2 WMMAs ----
    v16bf pa0 = lds_frag2(&sP[wave][ln16][sel * 8],
                          &sP[wave][ln16][16 + sel * 8]);
    v16bf pa1 = lds_frag2(&sP[wave][16 + ln16][sel * 8],
                          &sP[wave][16 + ln16][16 + sel * 8]);
    #pragma unroll
    for (int dt = 0; dt < 8; ++dt) {
      v16bf vb = lds_frag2(&sVt[buf][dt * 16 + ln16][sel * 8],
                           &sVt[buf][dt * 16 + ln16][16 + sel * 8]);
      o[0][dt] = __builtin_amdgcn_wmma_f32_16x16x32_bf16(
                     false, pa0, false, vb, (short)0, o[0][dt], false, false);
      o[1][dt] = __builtin_amdgcn_wmma_f32_16x16x32_bf16(
                     false, pa1, false, vb, (short)0, o[1][dt], false, false);
    }
  }

  // ---- epilogue: normalize by row sums, store fp32 ----
  #pragma unroll
  for (int mt = 0; mt < 2; ++mt) {
    #pragma unroll
    for (int r = 0; r < 8; ++r) {
      float inv = 1.0f / lrow[mt][r];
      const int m = mt * 16 + r + sel * 8;
      float* op = out + (((size_t)b * Ss + q0 + m) * Hh + h) * Dd + ln16;
      #pragma unroll
      for (int dt = 0; dt < 8; ++dt)
        op[dt * 16] = o[mt][dt][r] * inv;
    }
  }
}

extern "C" void kernel_launch(void* const* d_in, const int* in_sizes, int n_in,
                              void* d_out, int out_size, void* d_ws, size_t ws_size,
                              hipStream_t stream) {
  (void)in_sizes; (void)n_in; (void)out_size; (void)d_ws; (void)ws_size;
  const float* q = (const float*)d_in[0];
  const float* k = (const float*)d_in[1];
  const float* v = (const float*)d_in[2];
  float* out = (float*)d_out;

  dim3 grid(Bb * Hh * (Ss / BM));   // 512 workgroups
  dim3 block(128);                  // 4 wave32 waves, 32 Q rows each
  hipLaunchKernelGGL(fa_bf16_wmma_kernel, grid, block, 0, stream, q, k, v, out);
}
